// Transformer_enc_44049184587820
// MI455X (gfx1250) — compile-verified, matrix operand order checked by reading, not run on hardware
//
#include <hip/hip_runtime.h>
#include <hip/hip_bf16.h>

typedef __bf16 bf16;
typedef __attribute__((ext_vector_type(16))) __bf16 v16bf;
typedef __attribute__((ext_vector_type(8)))  __bf16 v8bf;
typedef __attribute__((ext_vector_type(8)))  float  v8f;

#define B_    64
#define S_    256
#define FIN_  64
#define D_    512
#define H_    8
#define DH_   64
#define DFF_  2048
#define L_    6
#define ROWS_ (B_ * S_)   // 16384
#define FEAT_ 32
#define ADIM_ 8
#define OUT_  5

// ---------------------------------------------------------------------------
// Fragment helpers (CDNA5 wave32 WMMA 16x16x32 bf16)
// 16-bit A 16x32 layout: lane L<16 -> row L, K chunks {0..7, 16..23};
// lane L>=16 -> row L-16, K chunks {8..15, 24..31}. B consumed as Bt [N,K]
// with identical addressing (column of B == row of Bt).
// ---------------------------------------------------------------------------
__device__ __forceinline__ v16bf make_frag(const bf16* p) {
  v8bf lo = *reinterpret_cast<const v8bf*>(p);
  v8bf hi = *reinterpret_cast<const v8bf*>(p + 16);
  return __builtin_shufflevector(lo, hi, 0,1,2,3,4,5,6,7,8,9,10,11,12,13,14,15);
}

__device__ __forceinline__ v8f wmma_bf16(v16bf a, v16bf b, v8f c) {
  return __builtin_amdgcn_wmma_f32_16x16x32_bf16(false, a, false, b, (short)0, c,
                                                 false, false);
}

// ---------------------------------------------------------------------------
// General WMMA GEMM: C[M,N] = A[M,K] @ Bt[N,K]^T with fused epilogues.
// Block = 256 threads = 8 waves; each wave owns a 16 x (NT*16) C tile.
// grid = (ceil(M/128), N/(NT*16)). NT is compile-time: branch-free hot loop.
// ---------------------------------------------------------------------------
template <int NT>
__global__ __launch_bounds__(256) void gemm_bf16_wmma(
    const bf16* __restrict__ A, int lda,
    const bf16* __restrict__ Bt, int ldb,
    const float* __restrict__ bias,       // [N] or null
    const float* __restrict__ residual,   // [M,N] or null
    const float* __restrict__ pes,        // [S_, N] add pes[row % S_][col] or null
    float* __restrict__ Cf,               // f32 out or null
    bf16* __restrict__ Cb,                // bf16 out or null
    int M, int N, int K, int relu, int vtrans) {
  const int lane = threadIdx.x & 31;
  const int wave = threadIdx.x >> 5;
  const int mBase = blockIdx.x * 128 + wave * 16;
  const int nBase = blockIdx.y * (NT * 16);
  if (mBase >= M) return;

  const int row16 = lane & 15;
  const int half  = lane >> 4;

  const bf16* aRow = A + (size_t)(mBase + row16) * lda + half * 8;
  const bf16* bRow[NT];
  #pragma unroll
  for (int t = 0; t < NT; ++t)
    bRow[t] = Bt + (size_t)(nBase + t * 16 + row16) * ldb + half * 8;

  v8f acc[NT];
  #pragma unroll
  for (int t = 0; t < NT; ++t) acc[t] = (v8f){0,0,0,0,0,0,0,0};

  #pragma unroll 2
  for (int k0 = 0; k0 < K; k0 += 32) {
    __builtin_prefetch(aRow + k0 + 128, 0, 1);  // global_prefetch_b8
    v16bf a = make_frag(aRow + k0);
    #pragma unroll
    for (int t = 0; t < NT; ++t)
      acc[t] = wmma_bf16(a, make_frag(bRow[t] + k0), acc[t]);
  }

  // C layout: VGPR r, lanes 0-15 -> (M=r, N=lane); lanes 16-31 -> (M=r+8, N=lane-16)
  const int colIn = lane & 15;
  const int rOff  = half * 8;
  #pragma unroll
  for (int t = 0; t < NT; ++t) {
    const int nb = nBase + t * 16;
    #pragma unroll
    for (int r = 0; r < 8; ++r) {
      const int row = mBase + rOff + r;
      const int col = nb + colIn;
      float v = acc[t][r];
      if (bias)     v += bias[col];
      if (pes)      v += pes[(size_t)(row & (S_ - 1)) * N + col];
      if (residual) v += residual[(size_t)row * N + col];
      if (relu)     v = v > 0.f ? v : 0.f;
      if (Cf) Cf[(size_t)row * N + col] = v;
      if (Cb) {
        if (vtrans) {
          // [B,S,H,dh] -> [B,H,dh,S]
          const int bb = row >> 8, ss = row & 255, hh = col >> 6, dd = col & 63;
          Cb[((((size_t)bb * H_ + hh) * DH_ + dd) << 8) + ss] = (bf16)v;
        } else {
          Cb[(size_t)row * N + col] = (bf16)v;
        }
      }
    }
  }
}

// ---------------------------------------------------------------------------
// Fused attention: scores = q k^T / 8, softmax, ctx = attn @ v.
// One workgroup per (b,h); 8 waves, each wave handles 32 query rows (2 tiles).
// Register pressure deliberately bounded: scores are produced 4 column-tiles
// at a time and parked in LDS (bf16); softmax runs out of LDS; the 1/sum
// normalization is folded into the ctx epilogue via ds_bpermute broadcast.
// ---------------------------------------------------------------------------
__global__ __launch_bounds__(256) void attention_wmma(
    const bf16* __restrict__ q,    // [ROWS_, D_]
    const bf16* __restrict__ k,    // [ROWS_, D_]
    const bf16* __restrict__ vT,   // [B_, H_, DH_, S_]
    bf16* __restrict__ ctx) {      // [ROWS_, D_]
  __shared__ bf16 smS[8][16 * S_];   // 8 waves x 8KB = 64KB

  const int lane = threadIdx.x & 31;
  const int wave = threadIdx.x >> 5;
  const int bh = blockIdx.x;            // 0..511
  const int b = bh >> 3, h = bh & 7;

  const bf16* qBase = q + (size_t)b * S_ * D_ + h * DH_;
  const bf16* kBase = k + (size_t)b * S_ * D_ + h * DH_;
  const bf16* vBase = vT + (size_t)bh * DH_ * S_;

  const int row16 = lane & 15;
  const int half  = lane >> 4;
  const int colIn = lane & 15;
  const int rOff  = half * 8;

  const bf16* kRow = kBase + (size_t)row16 * D_ + half * 8;
  const bf16* vRow = vBase + (size_t)row16 * S_ + half * 8;

  #pragma unroll 1
  for (int mt = 0; mt < 2; ++mt) {
    const int mBase = wave * 32 + mt * 16;

    // ---- phase A: scores (x0.125) -> LDS, 4 column-tiles per pass
    const bf16* aRow = qBase + (size_t)(mBase + row16) * D_ + half * 8;
    const v16bf a0 = make_frag(aRow);
    const v16bf a1 = make_frag(aRow + 32);
    #pragma unroll 1
    for (int g = 0; g < 4; ++g) {
      v8f acc[4];
      #pragma unroll
      for (int t = 0; t < 4; ++t) acc[t] = (v8f){0,0,0,0,0,0,0,0};
      #pragma unroll
      for (int t = 0; t < 4; ++t) {
        const bf16* bp = kRow + (size_t)(g * 4 + t) * 16 * D_;
        acc[t] = wmma_bf16(a0, make_frag(bp), acc[t]);
        acc[t] = wmma_bf16(a1, make_frag(bp + 32), acc[t]);
      }
      #pragma unroll
      for (int t = 0; t < 4; ++t) {
        #pragma unroll
        for (int r = 0; r < 8; ++r)
          smS[wave][(rOff + r) * S_ + (g * 4 + t) * 16 + colIn] =
              (bf16)(acc[t][r] * 0.125f);
      }
    }
    __syncthreads();

    // ---- phase B: softmax in LDS. Lane pair (2r,2r+1) owns row r,
    // each lane scans 128 columns; exp() stored back unnormalized.
    {
      const int srow = lane >> 1;
      const int h2   = lane & 1;
      bf16* prow = &smS[wave][srow * S_ + h2 * 128];
      float m = -1e30f;
      #pragma unroll 4
      for (int c0 = 0; c0 < 128; c0 += 8) {
        v8bf ch = *reinterpret_cast<const v8bf*>(prow + c0);
        #pragma unroll
        for (int j = 0; j < 8; ++j) m = fmaxf(m, (float)ch[j]);
      }
      m = fmaxf(m, __shfl_xor(m, 1, 32));
      float s = 0.f;
      #pragma unroll 4
      for (int c0 = 0; c0 < 128; c0 += 8) {
        v8bf ch = *reinterpret_cast<const v8bf*>(prow + c0);
        v8bf eo;
        #pragma unroll
        for (int j = 0; j < 8; ++j) {
          float e = __expf((float)ch[j] - m);
          s += e;
          eo[j] = (bf16)e;
        }
        *reinterpret_cast<v8bf*>(prow + c0) = eo;
      }
      s += __shfl_xor(s, 1, 32);
      const float sinv = 1.f / s;
      __syncthreads();

      // ---- phase C: ctx tile [16,64] = attn[16,256] @ v[256,64]
      v8f cacc[4];
      #pragma unroll
      for (int nt = 0; nt < 4; ++nt) cacc[nt] = (v8f){0,0,0,0,0,0,0,0};
      const bf16* aLds = &smS[wave][row16 * S_ + half * 8];
      #pragma unroll 2
      for (int k0 = 0; k0 < S_; k0 += 32) {
        v16bf a = make_frag(aLds + k0);
        #pragma unroll
        for (int nt = 0; nt < 4; ++nt)
          cacc[nt] = wmma_bf16(a, make_frag(vRow + (size_t)nt * 16 * S_ + k0),
                               cacc[nt]);
      }
      #pragma unroll
      for (int nt = 0; nt < 4; ++nt) {
        #pragma unroll
        for (int r = 0; r < 8; ++r) {
          const float inv = __shfl(sinv, (rOff + r) << 1, 32);  // ds_bpermute
          const int row = b * S_ + mBase + rOff + r;
          const int col = h * DH_ + nt * 16 + colIn;
          ctx[(size_t)row * D_ + col] = (bf16)(cacc[nt][r] * inv);
        }
      }
    }
    __syncthreads();
  }
}

// ---------------------------------------------------------------------------
// BatchNorm (training-mode batch stats over all B*S rows, per channel)
// ---------------------------------------------------------------------------
__global__ __launch_bounds__(256) void bn_stats(const float* __restrict__ y,
                                                float* __restrict__ mu,
                                                float* __restrict__ rinv) {
  const int c = blockIdx.x;
  float s = 0.f, s2 = 0.f;
  for (int r = threadIdx.x; r < ROWS_; r += 256) {
    float v = y[(size_t)r * D_ + c];
    s += v; s2 += v * v;
  }
  __shared__ float sh[512];
  sh[threadIdx.x] = s;
  sh[256 + threadIdx.x] = s2;
  __syncthreads();
  for (int off = 128; off > 0; off >>= 1) {
    if (threadIdx.x < off) {
      sh[threadIdx.x] += sh[threadIdx.x + off];
      sh[256 + threadIdx.x] += sh[256 + threadIdx.x + off];
    }
    __syncthreads();
  }
  if (threadIdx.x == 0) {
    float m = sh[0] / (float)ROWS_;
    float var = sh[256] / (float)ROWS_ - m * m;
    mu[c] = m;
    rinv[c] = rsqrtf(var + 1e-3f);
  }
}

__global__ void bn_apply(const float* __restrict__ y, const float* __restrict__ mu,
                         const float* __restrict__ rinv,
                         const float* __restrict__ gamma,
                         const float* __restrict__ beta,
                         float* __restrict__ outf, bf16* __restrict__ outb) {
  size_t i = (size_t)blockIdx.x * blockDim.x + threadIdx.x;
  if (i >= (size_t)ROWS_ * D_) return;
  const int c = (int)(i & (D_ - 1));
  float v = gamma[c] * (y[i] - mu[c]) * rinv[c] + beta[c];
  outf[i] = v;
  outb[i] = (bf16)v;
}

// ---------------------------------------------------------------------------
// Per-row LayerNorm (writes f32 + bf16 copies for the next GEMM)
// ---------------------------------------------------------------------------
__global__ __launch_bounds__(256) void layernorm_k(const float* __restrict__ z,
                                                   const float* __restrict__ gamma,
                                                   const float* __restrict__ beta,
                                                   float* __restrict__ outf,
                                                   bf16* __restrict__ outb) {
  const int row = blockIdx.x;
  const float* zr = z + (size_t)row * D_;
  float s = 0.f, s2 = 0.f;
  for (int c = threadIdx.x; c < D_; c += 256) {
    float v = zr[c]; s += v; s2 += v * v;
  }
  __shared__ float sh[512];
  sh[threadIdx.x] = s;
  sh[256 + threadIdx.x] = s2;
  __syncthreads();
  for (int off = 128; off > 0; off >>= 1) {
    if (threadIdx.x < off) {
      sh[threadIdx.x] += sh[threadIdx.x + off];
      sh[256 + threadIdx.x] += sh[256 + threadIdx.x + off];
    }
    __syncthreads();
  }
  const float m = sh[0] * (1.f / (float)D_);
  const float var = sh[256] * (1.f / (float)D_) - m * m;
  const float rs = rsqrtf(var + 1e-3f);
  for (int c = threadIdx.x; c < D_; c += 256) {
    float v = (zr[c] - m) * rs * gamma[c] + beta[c];
    outf[(size_t)row * D_ + c] = v;
    outb[(size_t)row * D_ + c] = (bf16)v;
  }
}

// ---------------------------------------------------------------------------
// Prep kernels
// ---------------------------------------------------------------------------
__global__ void transpose_to_bf16(const float* __restrict__ src,   // [K,N]
                                  bf16* __restrict__ dst,          // [N,K]
                                  int K, int N) {
  size_t i = (size_t)blockIdx.x * blockDim.x + threadIdx.x;
  if (i >= (size_t)K * N) return;
  int kk = (int)(i / N), n = (int)(i % N);
  dst[(size_t)n * K + kk] = (bf16)src[i];
}

__global__ void to_bf16_k(const float* __restrict__ s, bf16* __restrict__ d, int n) {
  int i = blockIdx.x * blockDim.x + threadIdx.x;
  if (i < n) d[i] = (bf16)s[i];
}

// ---------------------------------------------------------------------------
// Head: out = concat(feat, action) @ Wfin + bfin
// ---------------------------------------------------------------------------
__global__ void head_k(const float* __restrict__ feat, const float* __restrict__ action,
                       const float* __restrict__ Wfin, const float* __restrict__ bfin,
                       float* __restrict__ out) {
  int i = blockIdx.x * blockDim.x + threadIdx.x;
  if (i >= B_ * OUT_) return;
  int b = i / OUT_, o = i % OUT_;
  float acc = bfin[o];
  for (int j = 0; j < FEAT_; ++j) acc += feat[b * FEAT_ + j] * Wfin[j * OUT_ + o];
  for (int j = 0; j < ADIM_; ++j) acc += action[b * ADIM_ + j] * Wfin[(FEAT_ + j) * OUT_ + o];
  out[i] = acc;
}

// ---------------------------------------------------------------------------
static inline void* ws_take(char*& p, size_t bytes) {
  void* r = (void*)p;
  p += (bytes + 255) & ~(size_t)255;
  return r;
}

extern "C" void kernel_launch(void* const* d_in, const int* in_sizes, int n_in,
                              void* d_out, int out_size, void* d_ws, size_t ws_size,
                              hipStream_t stream) {
  const float* state  = (const float*)d_in[0];
  const float* action = (const float*)d_in[1];
  const float* pes    = (const float*)d_in[2];
  const float* W_emb  = (const float*)d_in[3];
  const float* b_emb  = (const float*)d_in[4];
  const float* Wq     = (const float*)d_in[5];
  const float* bq     = (const float*)d_in[6];
  const float* Wk     = (const float*)d_in[7];
  const float* bk     = (const float*)d_in[8];
  const float* Wv     = (const float*)d_in[9];
  const float* bv     = (const float*)d_in[10];
  const float* Wo     = (const float*)d_in[11];
  const float* bo     = (const float*)d_in[12];
  const float* bn_g   = (const float*)d_in[13];
  const float* bn_b   = (const float*)d_in[14];
  const float* W1     = (const float*)d_in[15];
  const float* b1     = (const float*)d_in[16];
  const float* W2     = (const float*)d_in[17];
  const float* b2     = (const float*)d_in[18];
  const float* ln_g   = (const float*)d_in[19];
  const float* ln_b   = (const float*)d_in[20];
  const float* Wf     = (const float*)d_in[21];
  const float* bfv    = (const float*)d_in[22];
  const float* Wfin   = (const float*)d_in[23];
  const float* bfin   = (const float*)d_in[24];

  // ---- workspace layout
  char* p = (char*)d_ws;
  float* xf    = (float*)ws_take(p, (size_t)ROWS_ * D_ * 4);
  bf16*  xb    = (bf16*) ws_take(p, (size_t)ROWS_ * D_ * 2);
  bf16*  qb    = (bf16*) ws_take(p, (size_t)ROWS_ * D_ * 2);
  bf16*  kb    = (bf16*) ws_take(p, (size_t)ROWS_ * D_ * 2);
  bf16*  vT    = (bf16*) ws_take(p, (size_t)ROWS_ * D_ * 2);
  bf16*  ctxb  = (bf16*) ws_take(p, (size_t)ROWS_ * D_ * 2);
  float* yf    = (float*)ws_take(p, (size_t)ROWS_ * D_ * 4);  // also z buffer
  float* ybnf  = (float*)ws_take(p, (size_t)ROWS_ * D_ * 4);
  bf16*  ybnb  = (bf16*) ws_take(p, (size_t)ROWS_ * D_ * 2);
  bf16*  h1b   = (bf16*) ws_take(p, (size_t)ROWS_ * DFF_ * 2);
  bf16*  stb   = (bf16*) ws_take(p, (size_t)ROWS_ * FIN_ * 2);
  float* mu    = (float*)ws_take(p, D_ * 4);
  float* rinv  = (float*)ws_take(p, D_ * 4);
  float* featf = (float*)ws_take(p, B_ * FEAT_ * 4);
  bf16*  WembT = (bf16*) ws_take(p, (size_t)D_ * FIN_ * 2);
  bf16*  WqT   = (bf16*) ws_take(p, (size_t)L_ * D_ * D_ * 2);
  bf16*  WkT   = (bf16*) ws_take(p, (size_t)L_ * D_ * D_ * 2);
  bf16*  WvT   = (bf16*) ws_take(p, (size_t)L_ * D_ * D_ * 2);
  bf16*  WoT   = (bf16*) ws_take(p, (size_t)L_ * D_ * D_ * 2);
  bf16*  W1T   = (bf16*) ws_take(p, (size_t)L_ * D_ * DFF_ * 2);
  bf16*  W2T   = (bf16*) ws_take(p, (size_t)L_ * DFF_ * D_ * 2);
  bf16*  WfT   = (bf16*) ws_take(p, (size_t)S_ * D_ * FEAT_ * 2);

  const dim3 blk(256);

  // ---- prep: conversions / weight transposes
  {
    int tot = ROWS_ * FIN_;
    to_bf16_k<<<dim3((tot + 255) / 256), blk, 0, stream>>>(state, stb, tot);
    transpose_to_bf16<<<dim3((FIN_ * D_ + 255) / 256), blk, 0, stream>>>(W_emb, WembT, FIN_, D_);
    for (int l = 0; l < L_; ++l) {
      size_t od = (size_t)l * D_ * D_;
      size_t of = (size_t)l * D_ * DFF_;
      transpose_to_bf16<<<dim3((D_ * D_ + 255) / 256), blk, 0, stream>>>(Wq + od, WqT + od, D_, D_);
      transpose_to_bf16<<<dim3((D_ * D_ + 255) / 256), blk, 0, stream>>>(Wk + od, WkT + od, D_, D_);
      transpose_to_bf16<<<dim3((D_ * D_ + 255) / 256), blk, 0, stream>>>(Wv + od, WvT + od, D_, D_);
      transpose_to_bf16<<<dim3((D_ * D_ + 255) / 256), blk, 0, stream>>>(Wo + od, WoT + od, D_, D_);
      transpose_to_bf16<<<dim3((D_ * DFF_ + 255) / 256), blk, 0, stream>>>(W1 + of, W1T + of, D_, DFF_);
      transpose_to_bf16<<<dim3((D_ * DFF_ + 255) / 256), blk, 0, stream>>>(W2 + of, W2T + of, DFF_, D_);
    }
    size_t wfe = (size_t)S_ * D_ * FEAT_;
    transpose_to_bf16<<<dim3((unsigned)((wfe + 255) / 256)), blk, 0, stream>>>(Wf, WfT, S_ * D_, FEAT_);
  }

  // ---- embedding: x = state @ W_emb + b_emb + pes
  gemm_bf16_wmma<4><<<dim3(ROWS_ / 128, D_ / 64), blk, 0, stream>>>(
      stb, FIN_, WembT, FIN_, b_emb, nullptr, pes, xf, xb,
      ROWS_, D_, FIN_, 0, 0);

  // ---- encoder layers
  const dim3 gD(ROWS_ / 128, D_ / 64);     // N=512
  const dim3 gF(ROWS_ / 128, DFF_ / 64);   // N=2048
  for (int l = 0; l < L_; ++l) {
    const size_t od = (size_t)l * D_ * D_;
    const size_t of = (size_t)l * D_ * DFF_;
    gemm_bf16_wmma<4><<<gD, blk, 0, stream>>>(xb, D_, WqT + od, D_, bq + l * D_,
        nullptr, nullptr, nullptr, qb, ROWS_, D_, D_, 0, 0);
    gemm_bf16_wmma<4><<<gD, blk, 0, stream>>>(xb, D_, WkT + od, D_, bk + l * D_,
        nullptr, nullptr, nullptr, kb, ROWS_, D_, D_, 0, 0);
    gemm_bf16_wmma<4><<<gD, blk, 0, stream>>>(xb, D_, WvT + od, D_, bv + l * D_,
        nullptr, nullptr, nullptr, vT, ROWS_, D_, D_, 0, 1);   // transposed store

    attention_wmma<<<dim3(B_ * H_), blk, 0, stream>>>(qb, kb, vT, ctxb);

    gemm_bf16_wmma<4><<<gD, blk, 0, stream>>>(ctxb, D_, WoT + od, D_, bo + l * D_,
        xf, nullptr, yf, nullptr, ROWS_, D_, D_, 0, 0);        // + residual(x)

    bn_stats<<<dim3(D_), blk, 0, stream>>>(yf, mu, rinv);
    bn_apply<<<dim3((ROWS_ * D_) / 256), blk, 0, stream>>>(
        yf, mu, rinv, bn_g + l * D_, bn_b + l * D_, ybnf, ybnb);

    gemm_bf16_wmma<4><<<gF, blk, 0, stream>>>(ybnb, D_, W1T + of, D_, b1 + l * DFF_,
        nullptr, nullptr, nullptr, h1b, ROWS_, DFF_, D_, 1, 0);  // ReLU
    gemm_bf16_wmma<4><<<gD, blk, 0, stream>>>(h1b, DFF_, W2T + of, DFF_, b2 + l * D_,
        ybnf, nullptr, yf, nullptr, ROWS_, D_, DFF_, 0, 0);      // + residual(ybn)

    layernorm_k<<<dim3(ROWS_), blk, 0, stream>>>(
        yf, ln_g + l * D_, ln_b + l * D_, xf, xb);
  }

  // ---- feat = flat(enc) @ Wf + bf   (M=64, N=32, K=131072) -> NT=2, grid (1,1)
  gemm_bf16_wmma<2><<<dim3(1, 1), blk, 0, stream>>>(
      xb, S_ * D_, WfT, S_ * D_, bfv, nullptr, nullptr, featf, nullptr,
      B_, FEAT_, S_ * D_, 0, 0);

  // ---- out = concat(feat, action) @ Wfin + bfin
  head_k<<<dim3(2), blk, 0, stream>>>(featf, action, Wfin, bfin, (float*)d_out);
}
